// GATEncoder_65644280152318
// MI455X (gfx1250) — compile-verified, matrix operand order checked by reading, not run on hardware
//
#include <hip/hip_runtime.h>
#include <hip/hip_bf16.h>

#define NN   20000
#define EE   320000
#define ETOT (EE + NN)
#define GG   512
#define INF_ 74
#define KP1  96      // INF_ padded to multiple of 32
#define EDF  12
#define HIDF 64
#define OUTF 128
#define NH   4
#define HC1  (NH * HIDF)   // 256
#define HC2  (NH * OUTF)   // 512

typedef __attribute__((ext_vector_type(16))) _Float16 v16h;
typedef __attribute__((ext_vector_type(8)))  _Float16 v8h;
typedef __attribute__((ext_vector_type(8)))  float    v8f;

// ---------- helpers ----------
__device__ __forceinline__ unsigned ordKey(float f) {
    unsigned u = __float_as_uint(f);
    return (u & 0x80000000u) ? ~u : (u | 0x80000000u);
}
__device__ __forceinline__ float invOrdKey(unsigned k) {
    return (k & 0x80000000u) ? __uint_as_float(k & 0x7fffffffu)
                             : __uint_as_float(~k);
}

// ---------- generic zero fill ----------
__global__ void fill_zero(float* __restrict__ p, long n) {
    long i = (long)blockIdx.x * blockDim.x + threadIdx.x;
    if (i < n) p[i] = 0.f;
}

// ---------- pack fp32 [M,K] -> fp16 [M,Kp] (zero-padded K) ----------
__global__ void pack_a_f16(const float* __restrict__ A, _Float16* __restrict__ Ah,
                           long M, int K, int Kp) {
    long i = (long)blockIdx.x * blockDim.x + threadIdx.x;
    if (i >= M * Kp) return;
    long r = i / Kp;
    int  k = (int)(i - r * Kp);
    Ah[i] = (k < K) ? (_Float16)A[r * K + k] : (_Float16)0.f;
}

// ---------- pack fp32 B[K,N] -> fp16 Bt[N,Kp] (transposed, zero-padded) ----------
__global__ void pack_bt_f16(const float* __restrict__ B, _Float16* __restrict__ Bh,
                            int K, long N, int Kp) {
    long i = (long)blockIdx.x * blockDim.x + threadIdx.x;
    if (i >= N * Kp) return;
    long n = i / Kp;
    int  k = (int)(i - n * Kp);
    Bh[i] = (k < K) ? (_Float16)B[(long)k * N + n] : (_Float16)0.f;
}

// ---------- WMMA GEMM: C[M,N] = Ah[M,Kp] * Bh[N,Kp]^T ; one wave = 16x64 strip ----------
__global__ __launch_bounds__(256) void wmma_gemm(const _Float16* __restrict__ Ah,
                                                 const _Float16* __restrict__ Bh,
                                                 float* __restrict__ C,
                                                 int M, int Kp, int N) {
    int wave = blockIdx.x * (blockDim.x >> 5) + (threadIdx.x >> 5);
    int lane = threadIdx.x & 31;
    int ngrp = N >> 6;                       // 64-wide column groups
    int mt = wave / ngrp;
    int ng = wave - mt * ngrp;
    if (mt * 16 >= M) return;
    int mloc = lane & 15, half = lane >> 4;
    const _Float16* arow  = Ah + (long)(mt * 16 + mloc) * Kp;
    const _Float16* brow0 = Bh + (long)(ng * 64 + mloc) * Kp;

    v8f acc[4] = {v8f{}, v8f{}, v8f{}, v8f{}};
    for (int kb = 0; kb < Kp; kb += 32) {
        v8h alo = *(const v8h*)(arow + kb + 8 * half);
        v8h ahi = *(const v8h*)(arow + kb + 16 + 8 * half);
        v16h a = __builtin_shufflevector(alo, ahi, 0, 1, 2, 3, 4, 5, 6, 7,
                                         8, 9, 10, 11, 12, 13, 14, 15);
#pragma unroll
        for (int t = 0; t < 4; ++t) {
            const _Float16* brow = brow0 + (long)t * 16 * Kp;
            v8h blo = *(const v8h*)(brow + kb + 8 * half);
            v8h bhi = *(const v8h*)(brow + kb + 16 + 8 * half);
            v16h b = __builtin_shufflevector(blo, bhi, 0, 1, 2, 3, 4, 5, 6, 7,
                                             8, 9, 10, 11, 12, 13, 14, 15);
            acc[t] = __builtin_amdgcn_wmma_f32_16x16x32_f16(
                false, a, false, b, (short)0, acc[t], false, false);
        }
    }
    int cn = ng * 64 + mloc;
#pragma unroll
    for (int t = 0; t < 4; ++t) {
#pragma unroll
        for (int v = 0; v < 8; ++v) {
            int cm = mt * 16 + v + 8 * half;
            C[(long)cm * N + cn + t * 16] = acc[t][v];
        }
    }
}

// ---------- edge-attr column sums (LDS reduce, then 12 global atomics/block) ----------
__global__ void easum_kernel(const float* __restrict__ ea, float* __restrict__ ea_sum) {
    __shared__ float sdata[EDF];
    if (threadIdx.x < EDF) sdata[threadIdx.x] = 0.f;
    __syncthreads();
    float loc[EDF];
#pragma unroll
    for (int k = 0; k < EDF; ++k) loc[k] = 0.f;
    for (long e = (long)blockIdx.x * blockDim.x + threadIdx.x; e < EE;
         e += (long)gridDim.x * blockDim.x) {
#pragma unroll
        for (int k = 0; k < EDF; ++k) loc[k] += ea[e * EDF + k];
    }
#pragma unroll
    for (int k = 0; k < EDF; ++k) atomicAdd(&sdata[k], loc[k]);
    __syncthreads();
    if (threadIdx.x < EDF) atomicAdd(&ea_sum[threadIdx.x], sdata[threadIdx.x]);
}

// ---------- w_eff[k][h] = sum_c We[k, h*C+c] * ae[h*C+c] ----------
template <int C>
__global__ void weff_kernel(const float* __restrict__ We, const float* __restrict__ ae,
                            float* __restrict__ weff) {
    int i = blockIdx.x * blockDim.x + threadIdx.x;   // (k,h)
    if (i >= EDF * NH) return;
    int k = i >> 2, h = i & 3;
    float s = 0.f;
    for (int c = 0; c < C; ++c) s += We[(long)k * NH * C + h * C + c] * ae[h * C + c];
    weff[k * NH + h] = s;
}

// ---------- a_edge for self-loop edges: dot(mean(ea), w_eff[:,h]) ----------
__global__ void aself_kernel(const float* __restrict__ ea_sum,
                             const float* __restrict__ weff,
                             float* __restrict__ aself) {
    int h = threadIdx.x;
    if (h >= NH) return;
    const float invE = 1.0f / (float)EE;
    float s = 0.f;
#pragma unroll
    for (int k = 0; k < EDF; ++k) s += (ea_sum[k] * invE) * weff[k * NH + h];
    aself[h] = s;
}

// ---------- per-node attention dots: as[n,h], ad[n,h] ----------
template <int C>
__global__ void nodeatt_kernel(const float* __restrict__ hpre,
                               const float* __restrict__ a_s,
                               const float* __restrict__ a_d,
                               float* __restrict__ as_, float* __restrict__ ad_) {
    int i = blockIdx.x * blockDim.x + threadIdx.x;   // (n,h)
    if (i >= NN * NH) return;
    int n = i >> 2, h = i & 3;
    const float* hp = hpre + (long)n * NH * C + h * C;
    float s1 = 0.f, s2 = 0.f;
    for (int c = 0; c < C; ++c) {
        float v = hp[c];
        s1 += v * a_s[h * C + c];
        s2 += v * a_d[h * C + c];
    }
    as_[i] = s1;
    ad_[i] = s2;
}

// ---------- alpha = lrelu(as[src]+ad[dst]+aedge); atomic max per (dst,h) ----------
__global__ void alpha_kernel(const float* __restrict__ ea,
                             const int* __restrict__ src, const int* __restrict__ dst,
                             const float* __restrict__ as_, const float* __restrict__ ad_,
                             const float* __restrict__ weff, const float* __restrict__ aself,
                             float* __restrict__ alpha, unsigned* __restrict__ maxbuf) {
    int e = blockIdx.x * blockDim.x + threadIdx.x;
    if (e >= ETOT) return;
    int s, d;
    float aed[NH];
    if (e < EE) {
        s = src[e]; d = dst[e];
        float v0 = 0.f, v1 = 0.f, v2 = 0.f, v3 = 0.f;
#pragma unroll
        for (int k = 0; k < EDF; ++k) {
            float x = ea[(long)e * EDF + k];
            v0 += x * weff[k * NH + 0];
            v1 += x * weff[k * NH + 1];
            v2 += x * weff[k * NH + 2];
            v3 += x * weff[k * NH + 3];
        }
        aed[0] = v0; aed[1] = v1; aed[2] = v2; aed[3] = v3;
    } else {
        s = d = e - EE;
#pragma unroll
        for (int h = 0; h < NH; ++h) aed[h] = aself[h];
    }
#pragma unroll
    for (int h = 0; h < NH; ++h) {
        float a = as_[s * NH + h] + ad_[d * NH + h] + aed[h];
        a = (a > 0.f) ? a : 0.2f * a;
        alpha[(long)e * NH + h] = a;
        atomicMax(&maxbuf[d * NH + h], ordKey(a));
    }
}

// ---------- ex = exp(alpha - max[dst]); atomic sum per (dst,h); alpha := ex ----------
__global__ void expsum_kernel(const int* __restrict__ dst,
                              float* __restrict__ alpha,
                              const unsigned* __restrict__ maxbuf,
                              float* __restrict__ sumbuf) {
    int e = blockIdx.x * blockDim.x + threadIdx.x;
    if (e >= ETOT) return;
    int d = (e < EE) ? dst[e] : (e - EE);
#pragma unroll
    for (int h = 0; h < NH; ++h) {
        float m  = invOrdKey(maxbuf[d * NH + h]);
        float ex = __expf(alpha[(long)e * NH + h] - m);
        alpha[(long)e * NH + h] = ex;
        atomicAdd(&sumbuf[d * NH + h], ex);
    }
}

// ---------- weighted scatter: agg[dst] += h[src] * w ----------
template <int C>
__global__ void agg_kernel(const float* __restrict__ hpre,
                           const float* __restrict__ alpha,
                           const float* __restrict__ sumbuf,
                           const int* __restrict__ src, const int* __restrict__ dst,
                           float* __restrict__ agg) {
    int e = blockIdx.x;
    int s, d;
    if (e < EE) { s = src[e]; d = dst[e]; } else { s = d = e - EE; }
    const int HC = NH * C;
    for (int t = threadIdx.x; t < HC; t += blockDim.x) {
        int h = t / C;
        float w = alpha[(long)e * NH + h] / (sumbuf[d * NH + h] + 1e-16f);
        atomicAdd(&agg[(long)d * HC + t], hpre[(long)s * HC + t] * w);
    }
}

// ---------- layer-1 finalize: h1h = fp16(relu(agg + b1))  (GEMM2's A operand) ----------
__global__ void fin1_kernel(const float* __restrict__ agg, const float* __restrict__ b1,
                            _Float16* __restrict__ h1h) {
    long i = (long)blockIdx.x * blockDim.x + threadIdx.x;
    if (i >= (long)NN * HC1) return;
    float v = agg[i] + b1[i % HC1];
    h1h[i] = (_Float16)((v > 0.f) ? v : 0.f);
}

// ---------- layer-2 finalize: h2 = relu(mean_h(agg) + b2) -> d_out ----------
__global__ void fin2_kernel(const float* __restrict__ agg, const float* __restrict__ b2,
                            float* __restrict__ out) {
    long i = (long)blockIdx.x * blockDim.x + threadIdx.x;
    if (i >= (long)NN * OUTF) return;
    long n = i / OUTF;
    int  c = (int)(i - n * OUTF);
    const float* a = agg + n * HC2;
    float v = 0.25f * (a[c] + a[OUTF + c] + a[2 * OUTF + c] + a[3 * OUTF + c]) + b2[c];
    out[i] = (v > 0.f) ? v : 0.f;
}

// ---------- graph pooling ----------
__global__ void cnt_kernel(const int* __restrict__ batch, float* __restrict__ gcnt) {
    int n = blockIdx.x * blockDim.x + threadIdx.x;
    if (n < NN) atomicAdd(&gcnt[batch[n]], 1.f);
}
__global__ void pool_kernel(const float* __restrict__ h2, const int* __restrict__ batch,
                            float* __restrict__ gsum) {
    long i = (long)blockIdx.x * blockDim.x + threadIdx.x;
    if (i >= (long)NN * OUTF) return;
    long n = i / OUTF;
    int  c = (int)(i - n * OUTF);
    atomicAdd(&gsum[(long)batch[n] * OUTF + c], h2[i]);
}
__global__ void scale_kernel(float* __restrict__ gsum, const float* __restrict__ gcnt) {
    int i = blockIdx.x * blockDim.x + threadIdx.x;
    if (i >= GG * OUTF) return;
    gsum[i] /= fmaxf(gcnt[i / OUTF], 1.f);
}

// ======================================================================
extern "C" void kernel_launch(void* const* d_in, const int* in_sizes, int n_in,
                              void* d_out, int out_size, void* d_ws, size_t ws_size,
                              hipStream_t stream) {
    const float* x   = (const float*)d_in[0];
    const float* ea  = (const float*)d_in[1];
    const float* W1  = (const float*)d_in[2];
    const float* aS1 = (const float*)d_in[3];
    const float* aD1 = (const float*)d_in[4];
    const float* We1 = (const float*)d_in[5];
    const float* aE1 = (const float*)d_in[6];
    const float* b1  = (const float*)d_in[7];
    const float* W2  = (const float*)d_in[8];
    const float* aS2 = (const float*)d_in[9];
    const float* aD2 = (const float*)d_in[10];
    const float* We2 = (const float*)d_in[11];
    const float* aE2 = (const float*)d_in[12];
    const float* b2  = (const float*)d_in[13];
    const int*   ei  = (const int*)d_in[14];
    const int*   bat = (const int*)d_in[15];
    const int* srcp = ei;
    const int* dstp = ei + EE;
    float* out  = (float*)d_out;                // [N*OUT] h2, then [G*OUT] graph_emb
    float* gemb = out + (long)NN * OUTF;

    // workspace layout
    float* ws = (float*)d_ws;
    long off = 0;
    auto allocf = [&](long nf) { float* p = ws + off; off += (nf + 63) & ~63L; return p; };
    auto alloch = [&](long nh) { return (_Float16*)allocf((nh + 1) / 2); };

    float*    h1pre = allocf((long)NN * HC1);       // GEMM1 out
    float*    h2pre = allocf((long)NN * HC2);       // GEMM2 out
    float*    agg1  = allocf((long)NN * HC1);
    float*    agg2  = allocf((long)NN * HC2);
    float*    asbuf = allocf((long)NN * NH);
    float*    adbuf = allocf((long)NN * NH);
    float*    alphab= allocf((long)ETOT * NH);
    float*    maxb  = allocf((long)NN * NH);        // used as unsigned
    float*    sumb  = allocf((long)NN * NH);
    float*    weff  = allocf(EDF * NH);
    float*    aself = allocf(NH);
    float*    easum = allocf(EDF);
    float*    gcnt  = allocf(GG);
    _Float16* xh    = alloch((long)NN * KP1);       // x packed fp16 [N, 96]
    _Float16* W1h   = alloch((long)HC1 * KP1);      // W1^T packed  [256, 96]
    _Float16* h1h   = alloch((long)NN * HC1);       // relu(h1) fp16 [N, 256]
    _Float16* W2h   = alloch((long)HC2 * HC1);      // W2^T packed  [512, 256]

    const int T = 256;
    auto gs = [](long n, int t) { return (unsigned)((n + t - 1) / t); };

    // ---- zero accumulators (ws/d_out are poisoned / stale between calls) ----
    fill_zero<<<gs((long)NN * HC1, T), T, 0, stream>>>(agg1, (long)NN * HC1);
    fill_zero<<<gs((long)NN * HC2, T), T, 0, stream>>>(agg2, (long)NN * HC2);
    fill_zero<<<gs((long)NN * NH, T), T, 0, stream>>>(maxb, (long)NN * NH);
    fill_zero<<<gs((long)NN * NH, T), T, 0, stream>>>(sumb, (long)NN * NH);
    fill_zero<<<1, T, 0, stream>>>(easum, EDF);
    fill_zero<<<gs(GG, T), T, 0, stream>>>(gcnt, GG);
    fill_zero<<<gs((long)GG * OUTF, T), T, 0, stream>>>(gemb, (long)GG * OUTF);

    // ---- operand packing (fp32 -> fp16, pad K, transpose B) ----
    pack_a_f16<<<gs((long)NN * KP1, T), T, 0, stream>>>(x, xh, NN, INF_, KP1);
    pack_bt_f16<<<gs((long)HC1 * KP1, T), T, 0, stream>>>(W1, W1h, INF_, HC1, KP1);
    pack_bt_f16<<<gs((long)HC2 * HC1, T), T, 0, stream>>>(W2, W2h, HC1, HC2, HC1);

    // ---- shared edge preprocessing ----
    easum_kernel<<<256, T, 0, stream>>>(ea, easum);

    // =================== layer 1 ===================
    weff_kernel<HIDF><<<1, 64, 0, stream>>>(We1, aE1, weff);
    aself_kernel<<<1, 32, 0, stream>>>(easum, weff, aself);

    {   // h1pre = x @ W1    (M=20000, Kp=96, N=256): 16x64 strips
        long waves = (NN / 16) * (HC1 / 64);
        wmma_gemm<<<gs(waves, 8), 256, 0, stream>>>(xh, W1h, h1pre, NN, KP1, HC1);
    }
    nodeatt_kernel<HIDF><<<gs((long)NN * NH, T), T, 0, stream>>>(h1pre, aS1, aD1, asbuf, adbuf);
    alpha_kernel<<<gs(ETOT, T), T, 0, stream>>>(ea, srcp, dstp, asbuf, adbuf, weff, aself,
                                                alphab, (unsigned*)maxb);
    expsum_kernel<<<gs(ETOT, T), T, 0, stream>>>(dstp, alphab, (const unsigned*)maxb, sumb);
    agg_kernel<HIDF><<<ETOT, 256, 0, stream>>>(h1pre, alphab, sumb, srcp, dstp, agg1);
    fin1_kernel<<<gs((long)NN * HC1, T), T, 0, stream>>>(agg1, b1, h1h);

    // =================== layer 2 ===================
    fill_zero<<<gs((long)NN * NH, T), T, 0, stream>>>(maxb, (long)NN * NH);
    fill_zero<<<gs((long)NN * NH, T), T, 0, stream>>>(sumb, (long)NN * NH);
    weff_kernel<OUTF><<<1, 64, 0, stream>>>(We2, aE2, weff);
    aself_kernel<<<1, 32, 0, stream>>>(easum, weff, aself);

    {   // h2pre = h1 @ W2   (M=20000, Kp=256, N=512): 16x64 strips
        long waves = (NN / 16) * (HC2 / 64);
        wmma_gemm<<<gs(waves, 8), 256, 0, stream>>>(h1h, W2h, h2pre, NN, HC1, HC2);
    }
    nodeatt_kernel<OUTF><<<gs((long)NN * NH, T), T, 0, stream>>>(h2pre, aS2, aD2, asbuf, adbuf);
    alpha_kernel<<<gs(ETOT, T), T, 0, stream>>>(ea, srcp, dstp, asbuf, adbuf, weff, aself,
                                                alphab, (unsigned*)maxb);
    expsum_kernel<<<gs(ETOT, T), T, 0, stream>>>(dstp, alphab, (const unsigned*)maxb, sumb);
    agg_kernel<OUTF><<<ETOT, 256, 0, stream>>>(h2pre, alphab, sumb, srcp, dstp, agg2);
    fin2_kernel<<<gs((long)NN * OUTF, T), T, 0, stream>>>(agg2, b2, out);

    // =================== graph pooling ===================
    cnt_kernel<<<gs(NN, T), T, 0, stream>>>(bat, gcnt);
    pool_kernel<<<gs((long)NN * OUTF, T), T, 0, stream>>>(out, bat, gemb);
    scale_kernel<<<gs((long)GG * OUTF, T), T, 0, stream>>>(gemb, gcnt);
}
